// Attention_56281251447163
// MI455X (gfx1250) — matrix operand order repaired and checked by reading, not compiled, not run on hardware
//
#include <hip/hip_runtime.h>
#include <hip/hip_bf16.h>

typedef __attribute__((ext_vector_type(16))) __bf16 v16bf;
typedef __attribute__((ext_vector_type(8)))  float  v8f;

#define NH   12
#define HD   64
#define LQ   1024
#define CDIM 768
#define C3   2304
#define BATCH 4
#define BH   (BATCH * NH)   // 48

__device__ inline v8f wmma_bf16(v16bf a, v16bf b, v8f c) {
  return __builtin_amdgcn_wmma_f32_16x16x32_bf16(false, a, false, b, (short)0, c, false, false);
}

// ---------------------------------------------------------------- convert
__global__ void k_f32_to_bf16(const float* __restrict__ src, __bf16* __restrict__ dst, int n) {
  int i = blockIdx.x * blockDim.x + threadIdx.x;
  if (i < n) dst[i] = (__bf16)src[i];
}

// ---------------------------------------------------------------- GEMM: C(M,N) = A(MxK,bf16) @ W(NxK,bf16)^T + bias
// wave computes 32(M) x 64(N): 2 A-frags reuse each of 4 B-frags (8 WMMA / k-step)
// block = 4 waves stacked along M (128x64 block tile)
__global__ void k_gemm_bf16(const __bf16* __restrict__ A, const __bf16* __restrict__ W,
                            const float* __restrict__ bias, float* __restrict__ C,
                            int M, int N, int Kd) {
  const int lane = threadIdx.x & 31;
  const int wave = threadIdx.x >> 5;
  const int hio  = (lane >> 4) & 1;
  const int mr   = lane & 15;
  const int m0   = (blockIdx.x * 4 + wave) * 32;
  const int n0   = blockIdx.y * 64;

  v8f acc[2][4];
#pragma unroll
  for (int u = 0; u < 2; ++u)
#pragma unroll
    for (int t = 0; t < 4; ++t)
#pragma unroll
      for (int r = 0; r < 8; ++r) acc[u][t][r] = 0.f;

  for (int k0 = 0; k0 < Kd; k0 += 32) {
    // A fragments: rows m0+mr and m0+16+mr; K = 16*hio + {0..15}
    v16bf a[2];
#pragma unroll
    for (int u = 0; u < 2; ++u) {
      const __bf16* Ap = A + (size_t)(m0 + u * 16 + mr) * Kd + k0 + hio * 16;
#pragma unroll
      for (int i = 0; i < 8; ++i) { a[u][i] = Ap[i]; a[u][8 + i] = Ap[8 + i]; }
    }
    if (k0 + 32 < Kd) {
      __builtin_prefetch(A + (size_t)(m0 + mr) * Kd + k0 + 32, 0, 1);  // next A k-step
      __builtin_prefetch(W + (size_t)(n0 + mr) * Kd + k0 + 32, 0, 1);  // next B k-step
    }
#pragma unroll
    for (int t = 0; t < 4; ++t) {
      // B fragment: column n, K = 16*hio + {0..15}; B[k][n] = W[n][k]
      v16bf b;
      const __bf16* Wp = W + (size_t)(n0 + t * 16 + mr) * Kd + k0 + hio * 16;
#pragma unroll
      for (int i = 0; i < 16; ++i) b[i] = Wp[i];
      acc[0][t] = wmma_bf16(a[0], b, acc[0][t]);
      acc[1][t] = wmma_bf16(a[1], b, acc[1][t]);
    }
  }
#pragma unroll
  for (int u = 0; u < 2; ++u) {
#pragma unroll
    for (int t = 0; t < 4; ++t) {
      int n = n0 + t * 16 + mr;
      float bv = bias ? bias[n] : 0.f;
#pragma unroll
      for (int r = 0; r < 8; ++r) {
        int m = m0 + u * 16 + r + hio * 8;
        C[(size_t)m * N + n] = acc[u][t][r] + bv;
      }
    }
  }
}

// ---------------------------------------------------------------- RoPE + split qkv -> q/k bf16 (BH, L, HD), v^T bf16 (BH, HD, L)
__global__ void k_rope_split(const float* __restrict__ qkv,
                             __bf16* __restrict__ qb, __bf16* __restrict__ kb,
                             __bf16* __restrict__ vt) {
  size_t idx = (size_t)blockIdx.x * blockDim.x + threadIdx.x;   // BH*L*32
  if (idx >= (size_t)BH * LQ * 32) return;
  int p = idx & 31;            // channel pair
  size_t t = idx >> 5;
  int l = t & (LQ - 1);
  size_t bn = t >> 10;         // 0..47
  int nhead = (int)(bn % NH);
  int b = (int)(bn / NH);

  size_t base = ((size_t)(b * LQ + l)) * C3 + (size_t)nhead * HD + 2 * p;
  float q0 = qkv[base],            q1 = qkv[base + 1];
  float k0 = qkv[base + CDIM],     k1 = qkv[base + CDIM + 1];
  float v0 = qkv[base + 2 * CDIM], v1 = qkv[base + 2 * CDIM + 1];

  int band = p & 15;
  float freq = __expf(-0.5756462732485114f * (float)band);  // 10000^(-band/16)
  float tt = (p < 16) ? (float)(l & 31) : (float)(l >> 5);
  float ang = tt * freq;
  float cs = __cosf(ang), sn = __sinf(ang);
  float qr = q0 * cs - q1 * sn, qi = q0 * sn + q1 * cs;
  float kr = k0 * cs - k1 * sn, ki = k0 * sn + k1 * cs;

  size_t dst = (bn * LQ + l) * HD + 2 * p;
  qb[dst] = (__bf16)qr; qb[dst + 1] = (__bf16)qi;
  kb[dst] = (__bf16)kr; kb[dst + 1] = (__bf16)ki;
  // V stored transposed: vt[bh][d][l] so attention A-frags are contiguous
  vt[(bn * HD + 2 * p) * LQ + l]     = (__bf16)v0;
  vt[(bn * HD + 2 * p + 1) * LQ + l] = (__bf16)v1;
}

// ---------------------------------------------------------------- rel-pos bias tables via WMMA
// mode 0: for fixed (bh, h):  relh[w][k] = sum_c Q[h*32+w][c] * rph[(h-k+31)][c]   (q-row stride = HD)
// mode 1: for fixed (bh, w):  relw[h][k] = sum_c Q[h*32+w][c] * rpw[(w-k+31)][c]   (q-row stride = 32*HD)
// grid: (32, BH, 2), block = 32 (one wave computes the 32x32 output)
__global__ void k_relbias_wmma(const __bf16* __restrict__ qb,
                               const __bf16* __restrict__ rph_bf,
                               const __bf16* __restrict__ rpw_bf,
                               float* __restrict__ relh, float* __restrict__ relw) {
  const int lane = threadIdx.x & 31;
  const int hio  = (lane >> 4) & 1;
  const int nr   = lane & 15;
  const int idx  = blockIdx.x;   // h (mode 0) or w (mode 1)
  const int bh   = blockIdx.y;
  const int mode = blockIdx.z;

  const __bf16* tab = mode ? rpw_bf : rph_bf;
  const __bf16* Qh  = qb + (size_t)bh * LQ * HD;
  const size_t rowStride = mode ? (size_t)32 * HD : (size_t)HD;
  const size_t rowBase   = mode ? (size_t)idx * HD : (size_t)idx * 32 * HD;

  v8f acc[2][2];
#pragma unroll
  for (int mt = 0; mt < 2; ++mt)
#pragma unroll
    for (int nt = 0; nt < 2; ++nt)
#pragma unroll
      for (int r = 0; r < 8; ++r) acc[mt][nt][r] = 0.f;

#pragma unroll
  for (int c0 = 0; c0 < HD; c0 += 32) {
    v16bf a[2];
#pragma unroll
    for (int mt = 0; mt < 2; ++mt) {
      const __bf16* Ap = Qh + rowBase + (size_t)(mt * 16 + nr) * rowStride + c0 + hio * 16;
#pragma unroll
      for (int i = 0; i < 8; ++i) { a[mt][i] = Ap[i]; a[mt][8 + i] = Ap[8 + i]; }
    }
    v16bf bf[2];
#pragma unroll
    for (int nt = 0; nt < 2; ++nt) {
      int k = nt * 16 + nr;
      const __bf16* Bp = tab + (size_t)(idx - k + 31) * HD + c0 + hio * 16;
#pragma unroll
      for (int i = 0; i < 16; ++i) bf[nt][i] = Bp[i];
    }
#pragma unroll
    for (int mt = 0; mt < 2; ++mt)
#pragma unroll
      for (int nt = 0; nt < 2; ++nt) acc[mt][nt] = wmma_bf16(a[mt], bf[nt], acc[mt][nt]);
  }

  float* dst = mode ? relw : relh;
#pragma unroll
  for (int mt = 0; mt < 2; ++mt) {
#pragma unroll
    for (int nt = 0; nt < 2; ++nt) {
      int k = nt * 16 + nr;
#pragma unroll
      for (int r = 0; r < 8; ++r) {
        int m = mt * 16 + r + hio * 8;
        int l = mode ? (m * 32 + idx) : (idx * 32 + m);
        dst[((size_t)bh * LQ + l) * 32 + k] = acc[mt][nt][r];
      }
    }
  }
}

// ---------------------------------------------------------------- flash attention, one wave = 16 q rows
// Computes S^T tiles (A = K rows, B = Q^T), online softmax per lane column,
// accumulates O^T = V^T @ P^T (V^T pre-transposed -> contiguous b128 A-frag loads),
// writes bf16 attn output in (B, L, C) layout.
__global__ void k_attn(const __bf16* __restrict__ qg, const __bf16* __restrict__ kg,
                       const __bf16* __restrict__ vtg, const float* __restrict__ relh,
                       const float* __restrict__ relw, __bf16* __restrict__ attn_out) {
  const int lane = threadIdx.x & 31;
  const int wave = threadIdx.x >> 5;
  const int hio  = (lane >> 4) & 1;
  const int nq   = lane & 15;
  const int bh   = blockIdx.y;                    // 0..47
  const int q0   = (blockIdx.x * 4 + wave) * 16;  // grid.x = 16
  const int b    = bh / NH;
  const int nh_  = bh % NH;

  const __bf16* Q  = qg  + (size_t)bh * LQ * HD;
  const __bf16* K  = kg  + (size_t)bh * LQ * HD;
  const __bf16* VT = vtg + (size_t)bh * HD * LQ;   // [d][l]

  const int qrow = q0 + nq;
  const float* RHrow = relh + ((size_t)bh * LQ + qrow) * 32;
  const float* RWrow = relw + ((size_t)bh * LQ + qrow) * 32;

  // loop-invariant Q^T B-fragments (K-dim = channel, two halves of hd=64)
  v16bf bq0, bq1;
  {
    const __bf16* Qp = Q + (size_t)qrow * HD + hio * 16;
#pragma unroll
    for (int i = 0; i < 16; ++i) { bq0[i] = Qp[i]; bq1[i] = Qp[32 + i]; }
  }
  // loop-invariant rel_w contribution: kw = r + 8*hio (+16) regardless of j
  float rwE[8], rwO[8];
#pragma unroll
  for (int r = 0; r < 8; ++r) {
    rwE[r] = RWrow[r + 8 * hio];
    rwO[r] = RWrow[16 + r + 8 * hio];
  }

  v8f o[4];
#pragma unroll
  for (int t = 0; t < 4; ++t)
#pragma unroll
    for (int r = 0; r < 8; ++r) o[t][r] = 0.f;
  float mrun = -1e30f, lrun = 0.f;
  const float scale = 0.125f;  // 1/sqrt(64)

  for (int j = 0; j < 32; ++j) {          // 32 steps of 32 keys
    const int kb32 = j * 32;
    // ---- scores S^T for two 16-key tiles
    v8f sE, sO;
#pragma unroll
    for (int r = 0; r < 8; ++r) { sE[r] = 0.f; sO[r] = 0.f; }
    {
      const __bf16* KpE = K + (size_t)(kb32 + nq) * HD + hio * 16;
      const __bf16* KpO = KpE + 16 * HD;
      v16bf aE0, aE1, aO0, aO1;
#pragma unroll
      for (int i = 0; i < 8; ++i) {
        aE0[i] = KpE[i];      aE0[8 + i] = KpE[8 + i];
        aE1[i] = KpE[32 + i]; aE1[8 + i] = KpE[40 + i];
        aO0[i] = KpO[i];      aO0[8 + i] = KpO[8 + i];
        aO1[i] = KpO[32 + i]; aO1[8 + i] = KpO[40 + i];
      }
      sE = wmma_bf16(aE0, bq0, sE); sE = wmma_bf16(aE1, bq1, sE);
      sO = wmma_bf16(aO0, bq0, sO); sO = wmma_bf16(aO1, bq1, sO);
    }
    // ---- scale + decomposed rel-pos bias: kh == j for the whole tile, kw is per-slot constant
    const float rhj = RHrow[j];
#pragma unroll
    for (int r = 0; r < 8; ++r) {
      sE[r] = sE[r] * scale + rhj + rwE[r];
      sO[r] = sO[r] * scale + rhj + rwO[r];
    }
    // ---- online softmax stats per q column (lane pair l, l^16 share column nq)
    float tmax = -1e30f;
#pragma unroll
    for (int r = 0; r < 8; ++r) { tmax = fmaxf(tmax, sE[r]); tmax = fmaxf(tmax, sO[r]); }
    tmax = fmaxf(tmax, __shfl_xor(tmax, 16, 32));
    float mnew = fmaxf(mrun, tmax);
    float alpha = __expf(mrun - mnew);
    float psum = 0.f;
#pragma unroll
    for (int r = 0; r < 8; ++r) {
      sE[r] = __expf(sE[r] - mnew); psum += sE[r];
      sO[r] = __expf(sO[r] - mnew); psum += sO[r];
    }
    psum += __shfl_xor(psum, 16, 32);
    lrun = lrun * alpha + psum;
    mrun = mnew;
#pragma unroll
    for (int t = 0; t < 4; ++t)
#pragma unroll
      for (int r = 0; r < 8; ++r) o[t][r] *= alpha;
    // ---- convert C-layout P^T frags -> B-frag layout bf16 (lane-half swap via shfl)
    v16bf pb;
#pragma unroll
    for (int r = 0; r < 8; ++r) {
      float e  = sE[r], od = sO[r];
      float eX = __shfl_xor(e, 16, 32);
      float oX = __shfl_xor(od, 16, 32);
      float lowv  = hio ? oX : e;    // k = 16*hio + r   from lane nq
      float highv = hio ? od : eX;   // k = 16*hio + 8+r from lane nq+16
      pb[r]     = (__bf16)lowv;
      pb[8 + r] = (__bf16)highv;
    }
    // ---- O^T += V^T @ P^T  (4 d-tiles of 16); V^T rows contiguous in memory
#pragma unroll
    for (int t = 0; t < 4; ++t) {
      v16bf av;
      const __bf16* Vp = VT + (size_t)(t * 16 + nq) * LQ + kb32 + hio * 16;
#pragma unroll
      for (int i = 0; i < 8; ++i) { av[i] = Vp[i]; av[8 + i] = Vp[8 + i]; }
      o[t] = wmma_bf16(av, pb, o[t]);
    }
  }
  // ---- normalize and store bf16 into (B, L, C) layout for the projection GEMM
  float inv = 1.0f / lrun;
#pragma unroll
  for (int t = 0; t < 4; ++t) {
#pragma unroll
    for (int r = 0; r < 8; ++r) {
      int d = t * 16 + r + hio * 8;
      attn_out[((size_t)(b * LQ + qrow)) * CDIM + nh_ * HD + d] = (__bf16)(o[t][r] * inv);
    }
  }
}

// ---------------------------------------------------------------- launcher
extern "C" void kernel_launch(void* const* d_in, const int* in_sizes, int n_in,
                              void* d_out, int out_size, void* d_ws, size_t ws_size,
                              hipStream_t stream) {
  (void)in_sizes; (void)n_in; (void)out_size; (void)ws_size;
  const float* x      = (const float*)d_in[0];
  const float* qkv_w  = (const float*)d_in[1];
  const float* qkv_b  = (const float*)d_in[2];
  const float* proj_w = (const float*)d_in[3];
  const float* proj_b = (const float*)d_in[4];
  const float* rph    = (const float*)d_in[5];
  const float* rpw    = (const float*)d_in[6];
  float* out = (float*)d_out;

  char* ws = (char*)d_ws;
  size_t off = 0;
  auto alloc = [&](size_t bytes) -> void* {
    void* p = ws + off;
    off = (off + bytes + 255) & ~(size_t)255;
    return p;
  };

  const int M = BATCH * LQ;                       // 4096
  const int RELN = (2 * 32 - 1) * HD;             // 4032
  __bf16* x_bf    = (__bf16*)alloc((size_t)M * CDIM * 2);
  __bf16* qkvw_bf = (__bf16*)alloc((size_t)C3 * CDIM * 2);
  __bf16* projw_bf= (__bf16*)alloc((size_t)CDIM * CDIM * 2);
  __bf16* rph_bf  = (__bf16*)alloc((size_t)RELN * 2);
  __bf16* rpw_bf  = (__bf16*)alloc((size_t)RELN * 2);
  float*  qkv     = (float*)alloc((size_t)M * C3 * 4);
  __bf16* qb      = (__bf16*)alloc((size_t)BH * LQ * HD * 2);
  __bf16* kb      = (__bf16*)alloc((size_t)BH * LQ * HD * 2);
  __bf16* vt      = (__bf16*)alloc((size_t)BH * HD * LQ * 2);
  float*  relh    = (float*)alloc((size_t)BH * LQ * 32 * 4);
  float*  relw    = (float*)alloc((size_t)BH * LQ * 32 * 4);
  __bf16* attn_bf = (__bf16*)alloc((size_t)M * CDIM * 2);

  // 1) convert inputs to bf16
  {
    int n1 = M * CDIM, n2 = C3 * CDIM, n3 = CDIM * CDIM;
    k_f32_to_bf16<<<(n1 + 255) / 256, 256, 0, stream>>>(x, x_bf, n1);
    k_f32_to_bf16<<<(n2 + 255) / 256, 256, 0, stream>>>(qkv_w, qkvw_bf, n2);
    k_f32_to_bf16<<<(n3 + 255) / 256, 256, 0, stream>>>(proj_w, projw_bf, n3);
    k_f32_to_bf16<<<(RELN + 255) / 256, 256, 0, stream>>>(rph, rph_bf, RELN);
    k_f32_to_bf16<<<(RELN + 255) / 256, 256, 0, stream>>>(rpw, rpw_bf, RELN);
  }
  // 2) QKV GEMM: (4096 x 2304) = x_bf @ qkv_w^T + qkv_b
  k_gemm_bf16<<<dim3(M / 128, C3 / 64), 128, 0, stream>>>(x_bf, qkvw_bf, qkv_b, qkv, M, C3, CDIM);
  // 3) RoPE + split to q/k bf16 and transposed v
  {
    size_t n = (size_t)BH * LQ * 32;
    k_rope_split<<<(unsigned)((n + 255) / 256), 256, 0, stream>>>(qkv, qb, kb, vt);
  }
  // 4) rel-pos bias tables (WMMA)
  k_relbias_wmma<<<dim3(32, BH, 2), 32, 0, stream>>>(qb, rph_bf, rpw_bf, relh, relw);
  // 5) flash attention (WMMA), bf16 output in (B, L, C)
  k_attn<<<dim3(LQ / 64, BH), 128, 0, stream>>>(qb, kb, vt, relh, relw, attn_bf);
  // 6) output projection: out = attn_bf @ proj_w^T + proj_b
  k_gemm_bf16<<<dim3(M / 128, CDIM / 64), 128, 0, stream>>>(attn_bf, projw_bf, proj_b, out, M, CDIM, CDIM);
}
